// RRN_72507637891256
// MI455X (gfx1250) — matrix operand order repaired and checked by reading, not compiled, not run on hardware
//
#include <hip/hip_runtime.h>
#include <hip/hip_bf16.h>
#include <math.h>

// ---------------------------------------------------------------------------
// RRN (sudoku relational net) on MI455X / gfx1250.
// - All GEMMs on V_WMMA_F32_16X16X32_BF16 (wave32, fp32 accumulate).
// - Message MLP (85% of FLOPs) fused: edge gather -> 3 layers -> sum over 20
//   neighbors through LDS; no pair tensor ever touches HBM.
// - Activations (X, h, M, gin) live in HBM as bf16 (they are only consumed as
//   WMMA A operands), halving traffic and enabling raw TDM tile copies.
// - Uniform 2-D A tiles (g-MLP, LSTM, readout) are staged into LDS by the
//   Tensor Data Mover (tensor_load_to_lds + s_wait_tensorcnt), issued by wave0.
// - Weights pre-swizzled into the WMMA B-fragment layout: one contiguous 32B
//   load per fragment per lane.
// ---------------------------------------------------------------------------

typedef __bf16 bf16_t;
typedef __attribute__((ext_vector_type(16))) __bf16 v16bf;
typedef __attribute__((ext_vector_type(8)))  float  v8f;
typedef __attribute__((ext_vector_type(4)))  unsigned int v4u;
typedef __attribute__((ext_vector_type(4)))  int v4i;
typedef __attribute__((ext_vector_type(8)))  int v8i;

#define HID    96
#define NROWS  20736   /* BATCH(256) * NUM_NODES(81) */
#define NNODES 81
#define KEDGE  20
#define NDIG   9
#define RITERS 4

__device__ __forceinline__ v8f vzero8() {
  v8f z;
#pragma unroll
  for (int i = 0; i < 8; ++i) z[i] = 0.0f;
  return z;
}

__device__ __forceinline__ float sigmoidf_(float x) {
  return 1.0f / (1.0f + __expf(-x));
}

// ---------------------------------------------------------------------------
// TDM: DMA a contiguous (rows x 96) bf16 tile from global into LDS.
// Descriptor per CDNA5 ISA sec.8:
//   group0: count=1 | lds_addr | global_addr(57b) | type=2
//   group1: data_size=1(2B) | tensor_dim0=96 | tensor_dim1=rows |
//           tile_dim0=96 | tile_dim1=rows | tensor_dim0_stride=96
// Issue from one wave only; publish with s_wait_tensorcnt 0 + barrier.
// ---------------------------------------------------------------------------
__device__ __forceinline__ void tdm_load_tile(const bf16_t* gsrc, void* ldsDst,
                                              int rows) {
  unsigned long long ga = (unsigned long long)(const void*)gsrc;
  v4u g0;
  g0[0] = 1u;                                             // count=1 (valid)
  g0[1] = (unsigned)(unsigned long long)ldsDst;           // LDS byte offset
  g0[2] = (unsigned)ga;
  g0[3] = (unsigned)((ga >> 32) & 0x01FFFFFFu) | (2u << 30);  // type=2
  v8i g1;
  g1[0] = 0x00010000;                   // data_size = 1 (2 bytes)
  g1[1] = (int)(96u << 16);             // tensor_dim0[15:0] = 96
  g1[2] = (int)((unsigned)rows << 16);  // dim0 hi=0 | tensor_dim1 lo = rows
  g1[3] = (int)(96u << 16);             // dim1 hi=0 | tile_dim0 = 96
  g1[4] = rows;                         // tile_dim1 = rows (tile_dim2 = 0)
  g1[5] = 96;                           // tensor_dim0_stride = 96
  g1[6] = 0;
  g1[7] = 0;
#if __has_builtin(__builtin_amdgcn_tensor_load_to_lds)
  v4i z4 = {0, 0, 0, 0};
#if defined(__clang_major__) && (__clang_major__ >= 23)
  v8i z8 = {0, 0, 0, 0, 0, 0, 0, 0};
  __builtin_amdgcn_tensor_load_to_lds(g0, g1, z4, z4, z8, 0);
#else
  __builtin_amdgcn_tensor_load_to_lds(g0, g1, z4, z4, 0);
#endif
#else
  asm volatile("tensor_load_to_lds %0, %1" ::"s"(g0), "s"(g1) : "memory");
#endif
}

__device__ __forceinline__ void tdm_wait() {
#if __has_builtin(__builtin_amdgcn_s_wait_tensorcnt)
  __builtin_amdgcn_s_wait_tensorcnt(0);
#else
  asm volatile("s_wait_tensorcnt 0x0" ::: "memory");
#endif
}

// ---------------------------------------------------------------------------
// Wave-level GEMM: C(16 x NT*16) += A(16 x kTiles*32) * B.
// A: row-major bf16 in LDS. B: pre-swizzled fragment buffer in global
// (fragment f = ((kt+ktOff)*nTilesTot + nt), lane-major, 16 bf16 per lane).
// A-fragment layout per ISA 7.12.2 (16-bit A 16x32).
// ktOff lets a K-split A (two contiguous LDS blocks) accumulate in place.
// ---------------------------------------------------------------------------
template <int NT>
__device__ __forceinline__ void wave_gemm(const bf16_t* A, int ldA, int row0,
                                          const bf16_t* __restrict__ Wf,
                                          int kTiles, int nTilesTot, int nt0,
                                          int ktOff, v8f* acc) {
  const int lane = threadIdx.x & 31;
  const int rsel = lane & 15;
  const int hi8  = (lane >> 4) << 3;   // 0 or 8
  for (int kt = 0; kt < kTiles; ++kt) {
    const bf16_t* ap = A + (row0 + rsel) * ldA + (kt << 5) + hi8;
    v16bf a;
#pragma unroll
    for (int i = 0; i < 8; ++i) { a[i] = ap[i]; a[i + 8] = ap[16 + i]; }
#pragma unroll
    for (int nt = 0; nt < NT; ++nt) {
      const v16bf b = *(const v16bf*)(
          Wf + ((size_t)(((kt + ktOff) * nTilesTot + nt0 + nt) * 32 + lane) << 4));
      acc[nt] = __builtin_amdgcn_wmma_f32_16x16x32_bf16(
          false, a, false, b, (short)0, acc[nt], false, false);
    }
  }
}

// C/D layout per ISA: VGPR r -> row (row0 + r + (lane>=16?8:0)), col = lane&15.
template <int NT, bool RELU>
__device__ __forceinline__ void epi_bf16(bf16_t* dst, int ldD, int row0, int nt0,
                                         const v8f* acc,
                                         const float* __restrict__ bias) {
  const int lane = threadIdx.x & 31;
  const int rsel = lane & 15;
  const int rb = row0 + ((lane >> 4) << 3);
#pragma unroll
  for (int nt = 0; nt < NT; ++nt) {
    const int col = (nt0 + nt) * 16 + rsel;
    const float bv = bias[col];
#pragma unroll
    for (int r = 0; r < 8; ++r) {
      float v = acc[nt][r] + bv;
      if (RELU) v = fmaxf(v, 0.0f);
      dst[(rb + r) * ldD + col] = (bf16_t)v;
    }
  }
}

template <int NT, bool RELU>
__device__ __forceinline__ void epi_f32(float* dst, int ldD, int row0, int nt0,
                                        const v8f* acc,
                                        const float* __restrict__ bias) {
  const int lane = threadIdx.x & 31;
  const int rsel = lane & 15;
  const int rb = row0 + ((lane >> 4) << 3);
#pragma unroll
  for (int nt = 0; nt < NT; ++nt) {
    const int col = (nt0 + nt) * 16 + rsel;
    const float bv = bias[col];
#pragma unroll
    for (int r = 0; r < 8; ++r) {
      float v = acc[nt][r] + bv;
      if (RELU) v = fmaxf(v, 0.0f);
      dst[(rb + r) * ldD + col] = v;
    }
  }
}

// ---------------------------------------------------------------------------
// Weight repack: fp32 (dout, din) row-major -> bf16 WMMA B-fragments,
// lane-major (one contiguous 32B fragment slice per lane). Zero-pads K and N.
// ---------------------------------------------------------------------------
__global__ void k_pack_w(bf16_t* __restrict__ dst, const float* __restrict__ w,
                         int din, int dout, int Kpart, int koff, int Np) {
  int idx = blockIdx.x * blockDim.x + threadIdx.x;
  int total = Kpart * Np;
  if (idx >= total) return;
  int k = idx / Np, n = idx % Np;
  float v = (k < din && n < dout) ? w[(size_t)n * din + k] : 0.0f;
  int gk = k + koff;
  int kt = gk >> 5, kk = gk & 31;
  int nt = n >> 4,  nn = n & 15;
  int lane = nn + ((kk & 8) << 1);
  int i = (kk & 7) + ((kk & 16) >> 1);
  int nTiles = Np >> 4;
  dst[((size_t)((kt * nTiles + nt) * 32 + lane) << 4) + i] = (bf16_t)v;
}

// ---------------------------------------------------------------------------
// Embed gather + input MLP3: 16(pad32) -> 96 -> 96 -> 96. Writes X and h(=X)
// as bf16. 64 rows/block, 4 waves.
// ---------------------------------------------------------------------------
__global__ void __launch_bounds__(128)
k_in_mlp(const float* __restrict__ embed, const int* __restrict__ grids,
         const bf16_t* __restrict__ w0, const float* __restrict__ b0,
         const bf16_t* __restrict__ w1, const float* __restrict__ b1,
         const bf16_t* __restrict__ w2, const float* __restrict__ b2,
         bf16_t* __restrict__ X, bf16_t* __restrict__ h) {
  __shared__ __align__(16) bf16_t sA[64 * 32];
  __shared__ __align__(16) bf16_t sT1[64 * HID];
  __shared__ __align__(16) bf16_t sT2[64 * HID];
  const int tid = threadIdx.x;
  const int base = blockIdx.x * 64;
  {
    int r = tid >> 1, half = tid & 1;
    const float* e = embed + (size_t)grids[base + r] * 16;
#pragma unroll
    for (int c = 0; c < 16; ++c)
      sA[r * 32 + half * 16 + c] = half ? (bf16_t)0.0f : (bf16_t)e[c];
  }
  if (tid == 0) __builtin_prefetch(w0, 0, 1);
  __syncthreads();
  const int wv = tid >> 5, row0 = wv * 16;
  v8f acc[6];
#pragma unroll
  for (int i = 0; i < 6; ++i) acc[i] = vzero8();
  wave_gemm<6>(sA, 32, row0, w0, 1, 6, 0, 0, acc);
  epi_bf16<6, true>(sT1, HID, row0, 0, acc, b0);
  __syncthreads();
#pragma unroll
  for (int i = 0; i < 6; ++i) acc[i] = vzero8();
  wave_gemm<6>(sT1, HID, row0, w1, 3, 6, 0, 0, acc);
  epi_bf16<6, true>(sT2, HID, row0, 0, acc, b1);
  __syncthreads();
#pragma unroll
  for (int i = 0; i < 6; ++i) acc[i] = vzero8();
  wave_gemm<6>(sT2, HID, row0, w2, 3, 6, 0, 0, acc);
  {
    const int lane = tid & 31, rsel = lane & 15;
    const int rb = row0 + ((lane >> 4) << 3);
#pragma unroll
    for (int nt = 0; nt < 6; ++nt) {
      int col = nt * 16 + rsel;
      float bv = b2[col];
#pragma unroll
      for (int r = 0; r < 8; ++r) {
        float v = acc[nt][r] + bv;
        size_t off = (size_t)(base + rb + r) * HID + col;
        X[off] = (bf16_t)v;
        h[off] = (bf16_t)v;
      }
    }
  }
}

// ---------------------------------------------------------------------------
// Fused message pass: gather [h_self | h_neighbor] (192) -> MLP3 -> sum over
// 20 edges. 80 rows = 4 complete nodes per block, 5 waves. A0 (80x192 bf16)
// aliases the 80x96 f32 partial-output buffer (both 30KB).
// ---------------------------------------------------------------------------
__global__ void __launch_bounds__(160)
k_msg(const bf16_t* __restrict__ h, const int* __restrict__ edges,
      const bf16_t* __restrict__ w0, const float* __restrict__ b0,
      const bf16_t* __restrict__ w1, const float* __restrict__ b1,
      const bf16_t* __restrict__ w2, const float* __restrict__ b2,
      bf16_t* __restrict__ Mout) {
  __shared__ __align__(16) float  sOut[80 * HID];   // aliased as A0
  __shared__ __align__(16) bf16_t sT1[80 * HID];
  __shared__ __align__(16) bf16_t sT2[80 * HID];
  bf16_t* sA = (bf16_t*)sOut;
  const int tid = threadIdx.x;
  const long base = (long)blockIdx.x * 80;
  {
    int r = tid >> 1, half = tid & 1;
    long grow = base + r;
    int gnode = (int)(grow / KEDGE), k = (int)(grow % KEDGE);
    int bb = gnode / NNODES, nloc = gnode % NNODES;
    const bf16_t* src;
    if (half == 0) {
      src = h + (size_t)gnode * HID;
    } else {
      int ne = edges[nloc * KEDGE + k];
      src = h + ((size_t)bb * NNODES + ne) * HID;
    }
    const uint4* s4 = (const uint4*)src;                 // 96 bf16 = 12 x 16B
    uint4* d4 = (uint4*)(sA + r * 192 + half * HID);
#pragma unroll
    for (int c = 0; c < 12; ++c) d4[c] = s4[c];
  }
  if (tid == 0) __builtin_prefetch(w0, 0, 1);
  __syncthreads();
  const int wv = tid >> 5, row0 = wv * 16;
  v8f acc[6];
#pragma unroll
  for (int i = 0; i < 6; ++i) acc[i] = vzero8();
  wave_gemm<6>(sA, 192, row0, w0, 6, 6, 0, 0, acc);
  epi_bf16<6, true>(sT1, HID, row0, 0, acc, b0);
  __syncthreads();                       // all GEMM0 reads of sA done here
#pragma unroll
  for (int i = 0; i < 6; ++i) acc[i] = vzero8();
  wave_gemm<6>(sT1, HID, row0, w1, 3, 6, 0, 0, acc);
  epi_bf16<6, true>(sT2, HID, row0, 0, acc, b1);
  __syncthreads();
#pragma unroll
  for (int i = 0; i < 6; ++i) acc[i] = vzero8();
  wave_gemm<6>(sT2, HID, row0, w2, 3, 6, 0, 0, acc);
  epi_f32<6, false>(sOut, HID, row0, 0, acc, b2);  // safe: A0 no longer read
  __syncthreads();
  for (int idx = tid; idx < 4 * HID; idx += 160) {
    int j = idx / HID, c = idx % HID;
    float s = 0.0f;
#pragma unroll
    for (int r = 0; r < KEDGE; ++r) s += sOut[(j * KEDGE + r) * HID + c];
    Mout[(size_t)(base / KEDGE + j) * HID + c] = (bf16_t)s;
  }
}

// ---------------------------------------------------------------------------
// g-MLP3 on concat(X, M): 192 -> 96 -> 96 -> 96. 64 rows/block, 4 waves.
// X and M tiles are uniform contiguous 64x96 bf16 tiles -> TDM loads.
// ---------------------------------------------------------------------------
__global__ void __launch_bounds__(128)
k_g(const bf16_t* __restrict__ X, const bf16_t* __restrict__ Mbuf,
    const bf16_t* __restrict__ w0, const float* __restrict__ b0,
    const bf16_t* __restrict__ w1, const float* __restrict__ b1,
    const bf16_t* __restrict__ w2, const float* __restrict__ b2,
    bf16_t* __restrict__ gin) {
  __shared__ __align__(16) bf16_t sX[64 * HID];
  __shared__ __align__(16) bf16_t sM[64 * HID];
  __shared__ __align__(16) bf16_t sT1[64 * HID];
  __shared__ __align__(16) bf16_t sT2[64 * HID];
  const int tid = threadIdx.x;
  const int base = blockIdx.x * 64;
  if ((tid >> 5) == 0) {               // wave 0 drives the Tensor Data Mover
    tdm_load_tile(X + (size_t)base * HID, sX, 64);
    tdm_load_tile(Mbuf + (size_t)base * HID, sM, 64);
    tdm_wait();
  }
  __syncthreads();
  const int wv = tid >> 5, row0 = wv * 16;
  v8f acc[6];
#pragma unroll
  for (int i = 0; i < 6; ++i) acc[i] = vzero8();
  wave_gemm<6>(sX, HID, row0, w0, 3, 6, 0, 0, acc);   // K 0..95  (X half)
  wave_gemm<6>(sM, HID, row0, w0, 3, 6, 0, 3, acc);   // K 96..191 (M half)
  epi_bf16<6, true>(sT1, HID, row0, 0, acc, b0);
  __syncthreads();
#pragma unroll
  for (int i = 0; i < 6; ++i) acc[i] = vzero8();
  wave_gemm<6>(sT1, HID, row0, w1, 3, 6, 0, 0, acc);
  epi_bf16<6, true>(sT2, HID, row0, 0, acc, b1);
  __syncthreads();
#pragma unroll
  for (int i = 0; i < 6; ++i) acc[i] = vzero8();
  wave_gemm<6>(sT2, HID, row0, w2, 3, 6, 0, 0, acc);
  epi_bf16<6, false>(gin + (size_t)base * HID, HID, row0, 0, acc, b2);
}

// ---------------------------------------------------------------------------
// LSTM: gates = [gin | h] (192) x Wcat (192x384) + bih + bhh, cell update.
// 32 rows/block, 8 waves (2 row-tiles x 4 column quarters). gin/h tiles via
// TDM. h is re-written bf16; c stays fp32.
// ---------------------------------------------------------------------------
__global__ void __launch_bounds__(256)
k_lstm(const bf16_t* __restrict__ gin, bf16_t* __restrict__ h,
       const bf16_t* __restrict__ wf, const float* __restrict__ bih,
       const float* __restrict__ bhh, const float* __restrict__ cin,
       float* __restrict__ cout) {
  __shared__ __align__(16) bf16_t sG0[32 * HID];
  __shared__ __align__(16) bf16_t sH0[32 * HID];
  __shared__ __align__(16) float  sG[32 * 384];
  const int tid = threadIdx.x;
  const int base = blockIdx.x * 32;
  if ((tid >> 5) == 0) {
    tdm_load_tile(gin + (size_t)base * HID, sG0, 32);
    tdm_load_tile(h + (size_t)base * HID, sH0, 32);
    tdm_wait();
  }
  if (tid == 0) __builtin_prefetch(wf, 0, 1);
  __syncthreads();
  const int wv = tid >> 5;
  const int row0 = (wv >> 2) * 16;
  const int nt0 = (wv & 3) * 6;
  v8f acc[6];
#pragma unroll
  for (int i = 0; i < 6; ++i) acc[i] = vzero8();
  wave_gemm<6>(sG0, HID, row0, wf, 3, 24, nt0, 0, acc);  // K 0..95  (gin)
  wave_gemm<6>(sH0, HID, row0, wf, 3, 24, nt0, 3, acc);  // K 96..191 (h)
  {
    const int lane = tid & 31, rsel = lane & 15;
    const int rb = row0 + ((lane >> 4) << 3);
#pragma unroll
    for (int nt = 0; nt < 6; ++nt) {
      int col = (nt0 + nt) * 16 + rsel;
      float bv = bih[col] + bhh[col];
#pragma unroll
      for (int r = 0; r < 8; ++r) sG[(rb + r) * 384 + col] = acc[nt][r] + bv;
    }
  }
  __syncthreads();
  for (int idx = tid; idx < 32 * HID; idx += 256) {
    int r = idx / HID, c = idx % HID;
    size_t row = (size_t)(base + r);
    float gi = sG[r * 384 + c];
    float gf = sG[r * 384 + HID + c];
    float gg = sG[r * 384 + 2 * HID + c];
    float go = sG[r * 384 + 3 * HID + c];
    float cp = cin[row * HID + c];
    float cn = sigmoidf_(gf) * cp + sigmoidf_(gi) * tanhf(gg);
    float hn = sigmoidf_(go) * tanhf(cn);
    cout[row * HID + c] = cn;
    h[row * HID + c]    = (bf16_t)hn;
  }
}

// ---------------------------------------------------------------------------
// Readout MLP3: 96 -> 96 -> 96 -> 9 (N padded to 16). 64 rows/block, 4 waves.
// h tile staged via TDM.
// ---------------------------------------------------------------------------
__global__ void __launch_bounds__(128)
k_r(const bf16_t* __restrict__ h,
    const bf16_t* __restrict__ w0, const float* __restrict__ b0,
    const bf16_t* __restrict__ w1, const float* __restrict__ b1,
    const bf16_t* __restrict__ w2, const float* __restrict__ b2,
    float* __restrict__ out) {
  __shared__ __align__(16) bf16_t sA[64 * HID];
  __shared__ __align__(16) bf16_t sT1[64 * HID];
  __shared__ __align__(16) bf16_t sT2[64 * HID];
  const int tid = threadIdx.x;
  const int base = blockIdx.x * 64;
  if ((tid >> 5) == 0) {
    tdm_load_tile(h + (size_t)base * HID, sA, 64);
    tdm_wait();
  }
  if (tid == 0) __builtin_prefetch(w0, 0, 1);
  __syncthreads();
  const int wv = tid >> 5, row0 = wv * 16;
  v8f acc[6];
#pragma unroll
  for (int i = 0; i < 6; ++i) acc[i] = vzero8();
  wave_gemm<6>(sA, HID, row0, w0, 3, 6, 0, 0, acc);
  epi_bf16<6, true>(sT1, HID, row0, 0, acc, b0);
  __syncthreads();
#pragma unroll
  for (int i = 0; i < 6; ++i) acc[i] = vzero8();
  wave_gemm<6>(sT1, HID, row0, w1, 3, 6, 0, 0, acc);
  epi_bf16<6, true>(sT2, HID, row0, 0, acc, b1);
  __syncthreads();
  v8f a3 = vzero8();
  wave_gemm<1>(sT2, HID, row0, w2, 3, 1, 0, 0, &a3);
  {
    const int lane = tid & 31, rsel = lane & 15;
    const int rb = row0 + ((lane >> 4) << 3);
    if (rsel < NDIG) {
      float bv = b2[rsel];
#pragma unroll
      for (int r = 0; r < 8; ++r)
        out[(size_t)(base + rb + r) * NDIG + rsel] = a3[r] + bv;
    }
  }
}

// ---------------------------------------------------------------------------
// Launch
// ---------------------------------------------------------------------------
extern "C" void kernel_launch(void* const* d_in, const int* in_sizes, int n_in,
                              void* d_out, int out_size, void* d_ws,
                              size_t ws_size, hipStream_t stream) {
  const float* embed   = (const float*)d_in[0];
  const float* in_w0   = (const float*)d_in[1];
  const float* in_b0   = (const float*)d_in[2];
  const float* in_w1   = (const float*)d_in[3];
  const float* in_b1   = (const float*)d_in[4];
  const float* in_w2   = (const float*)d_in[5];
  const float* in_b2   = (const float*)d_in[6];
  const float* f_w0    = (const float*)d_in[7];
  const float* f_b0    = (const float*)d_in[8];
  const float* f_w1    = (const float*)d_in[9];
  const float* f_b1    = (const float*)d_in[10];
  const float* f_w2    = (const float*)d_in[11];
  const float* f_b2    = (const float*)d_in[12];
  const float* g_w0    = (const float*)d_in[13];
  const float* g_b0    = (const float*)d_in[14];
  const float* g_w1    = (const float*)d_in[15];
  const float* g_b1    = (const float*)d_in[16];
  const float* g_w2    = (const float*)d_in[17];
  const float* g_b2    = (const float*)d_in[18];
  const float* wih     = (const float*)d_in[19];
  const float* whh     = (const float*)d_in[20];
  const float* bih     = (const float*)d_in[21];
  const float* bhh     = (const float*)d_in[22];
  const float* r_w0    = (const float*)d_in[23];
  const float* r_b0    = (const float*)d_in[24];
  const float* r_w1    = (const float*)d_in[25];
  const float* r_b1    = (const float*)d_in[26];
  const float* r_w2    = (const float*)d_in[27];
  const float* r_b2    = (const float*)d_in[28];
  const float* c0      = (const float*)d_in[29];
  const int*   grids   = (const int*)d_in[30];
  const int*   edges   = (const int*)d_in[31];
  // d_in[32] = iters (fixed at 4 by the reference constants)

  unsigned char* p = (unsigned char*)d_ws;
  auto take = [&](size_t bytes) {
    unsigned char* r = p;
    p += (bytes + 255) & ~(size_t)255;
    return r;
  };
  // Fragment buffer size = (Kpad/32)*(Npad/16)*512 bf16 elements.
  bf16_t* wf_in0 = (bf16_t*)take((size_t)1 * 6 * 512 * 2);
  bf16_t* wf_in1 = (bf16_t*)take((size_t)3 * 6 * 512 * 2);
  bf16_t* wf_in2 = (bf16_t*)take((size_t)3 * 6 * 512 * 2);
  bf16_t* wf_f0  = (bf16_t*)take((size_t)6 * 6 * 512 * 2);
  bf16_t* wf_f1  = (bf16_t*)take((size_t)3 * 6 * 512 * 2);
  bf16_t* wf_f2  = (bf16_t*)take((size_t)3 * 6 * 512 * 2);
  bf16_t* wf_g0  = (bf16_t*)take((size_t)6 * 6 * 512 * 2);
  bf16_t* wf_g1  = (bf16_t*)take((size_t)3 * 6 * 512 * 2);
  bf16_t* wf_g2  = (bf16_t*)take((size_t)3 * 6 * 512 * 2);
  bf16_t* wf_lst = (bf16_t*)take((size_t)6 * 24 * 512 * 2);
  bf16_t* wf_r0  = (bf16_t*)take((size_t)3 * 6 * 512 * 2);
  bf16_t* wf_r1  = (bf16_t*)take((size_t)3 * 6 * 512 * 2);
  bf16_t* wf_r2  = (bf16_t*)take((size_t)3 * 1 * 512 * 2);
  bf16_t* X    = (bf16_t*)take((size_t)NROWS * HID * 2);
  bf16_t* hbuf = (bf16_t*)take((size_t)NROWS * HID * 2);
  bf16_t* Mbuf = (bf16_t*)take((size_t)NROWS * HID * 2);
  bf16_t* gbuf = (bf16_t*)take((size_t)NROWS * HID * 2);
  float*  cbuf = (float*)take((size_t)NROWS * HID * 4);

  auto pack = [&](bf16_t* dst, const float* w, int din, int dout, int Kpart,
                  int koff, int Np) {
    int total = Kpart * Np;
    k_pack_w<<<(total + 255) / 256, 256, 0, stream>>>(dst, w, din, dout, Kpart,
                                                      koff, Np);
  };
  pack(wf_in0, in_w0, 16, 96, 32, 0, 96);     // K padded 16 -> 32
  pack(wf_in1, in_w1, 96, 96, 96, 0, 96);
  pack(wf_in2, in_w2, 96, 96, 96, 0, 96);
  pack(wf_f0, f_w0, 192, 96, 192, 0, 96);
  pack(wf_f1, f_w1, 96, 96, 96, 0, 96);
  pack(wf_f2, f_w2, 96, 96, 96, 0, 96);
  pack(wf_g0, g_w0, 192, 96, 192, 0, 96);
  pack(wf_g1, g_w1, 96, 96, 96, 0, 96);
  pack(wf_g2, g_w2, 96, 96, 96, 0, 96);
  pack(wf_lst, wih, 96, 384, 96, 0, 384);     // rows 0..95   of Wcat
  pack(wf_lst, whh, 96, 384, 96, 96, 384);    // rows 96..191 of Wcat
  pack(wf_r0, r_w0, 96, 96, 96, 0, 96);
  pack(wf_r1, r_w1, 96, 96, 96, 0, 96);
  pack(wf_r2, r_w2, 96, 9, 96, 0, 16);        // N padded 9 -> 16

  k_in_mlp<<<NROWS / 64, 128, 0, stream>>>(embed, grids, wf_in0, in_b0, wf_in1,
                                           in_b1, wf_in2, in_b2, X, hbuf);

  float* out = (float*)d_out;
  for (int it = 0; it < RITERS; ++it) {
    k_msg<<<(NROWS * KEDGE) / 80, 160, 0, stream>>>(
        hbuf, edges, wf_f0, f_b0, wf_f1, f_b1, wf_f2, f_b2, Mbuf);
    k_g<<<NROWS / 64, 128, 0, stream>>>(X, Mbuf, wf_g0, g_b0, wf_g1, g_b1,
                                        wf_g2, g_b2, gbuf);
    k_lstm<<<NROWS / 32, 256, 0, stream>>>(gbuf, hbuf, wf_lst, bih, bhh,
                                           it == 0 ? c0 : cbuf, cbuf);
    k_r<<<NROWS / 64, 128, 0, stream>>>(hbuf, wf_r0, r_b0, wf_r1, r_b1, wf_r2,
                                        r_b2, out + (size_t)it * NROWS * NDIG);
  }
}
